// MultiHeadAttentionEin_59545426592321
// MI455X (gfx1250) — matrix operand order repaired and checked by reading, not compiled, not run on hardware
//
#include <hip/hip_runtime.h>
#include <hip/hip_bf16.h>

// ---------------------------------------------------------------------------
// MHA forward for MI455X (gfx1250): bf16 WMMA (16x16x32), f32 accumulate,
// TDM (tensor_load_to_lds) double-buffered K/V staging in the attention loop.
// Shapes: b=2, t=2048, d=1024, n=16 heads, h=64.  SCALE = 1/8.
// ---------------------------------------------------------------------------

typedef __bf16 v16bf __attribute__((ext_vector_type(16)));
typedef __bf16 v8bf  __attribute__((ext_vector_type(8)));
typedef float  v8f   __attribute__((ext_vector_type(8)));
typedef unsigned int v4ui __attribute__((ext_vector_type(4)));
typedef int v4i  __attribute__((ext_vector_type(4)));
typedef int v8i  __attribute__((ext_vector_type(8)));

#define WMMA_BF16(a, b, c) \
  __builtin_amdgcn_wmma_f32_16x16x32_bf16(false, (a), false, (b), (short)0, (c), false, false)

static constexpr int B   = 2;
static constexpr int T   = 2048;
static constexpr int DM  = 1024;   // model dim
static constexpr int NH  = 16;     // heads
static constexpr int H   = 64;     // head dim
static constexpr int BT  = B * T;  // 4096 rows
static constexpr float SCALE = 0.125f; // 1/sqrt(64)

__device__ __forceinline__ v16bf combine8(v8bf lo, v8bf hi) {
  v16bf r;
#pragma unroll
  for (int i = 0; i < 8; ++i) { r[i] = lo[i]; r[i + 8] = hi[i]; }
  return r;
}

// --------------------- Tensor Data Mover (TDM) helper ----------------------
// 2-D tile load Global -> LDS.  Descriptor packing per CDNA5 ISA:
//  g0: count=1, lds_addr, global_addr(57b), type=2
//  g1: data_size=1 (2-byte elems), tensor_dim0/1, tile_dim0/1, dim0 stride
//  groups 2/3 zero (tile_dim2==0 -> 2-D tensor)
__device__ __forceinline__ void tdm_load_2d(const void* gptr, void* lptr,
                                            unsigned tile0, unsigned tile1,
                                            unsigned tdim0, unsigned tdim1,
                                            unsigned long long stride0) {
  unsigned long long ga = (unsigned long long)(uintptr_t)gptr;
  unsigned lds_off = (unsigned)(uintptr_t)lptr;   // LDS byte offset (addr[31:0])
  v4ui g0 = { 1u,                                  // count=1, user mode
              lds_off,
              (unsigned)ga,
              (unsigned)((ga >> 32) & 0x01FFFFFFu) | (2u << 30) };
  v8i g1 = { (int)(1u << 16),                               // data_size=2B
             (int)((tdim0 & 0xFFFFu) << 16),                // tensor_dim0 lo
             (int)(((tdim0 >> 16) & 0xFFFFu) | ((tdim1 & 0xFFFFu) << 16)),
             (int)(((tdim1 >> 16) & 0xFFFFu) | ((tile0 & 0xFFFFu) << 16)),
             (int)(tile1 & 0xFFFFu),                        // tile_dim1, tile_dim2=0
             (int)(unsigned)(stride0 & 0xFFFFFFFFull),      // dim0 stride lo32
             (int)(unsigned)((stride0 >> 32) & 0xFFFFull),  // dim0 stride hi16
             0 };
  v4i z4 = { 0, 0, 0, 0 };
#if __clang_major__ >= 23
  v8i z8 = { 0, 0, 0, 0, 0, 0, 0, 0 };
  __builtin_amdgcn_tensor_load_to_lds(g0, g1, z4, z4, z8, 0);
#else
  __builtin_amdgcn_tensor_load_to_lds(g0, g1, z4, z4, 0);
#endif
}

// -------------------------- conversion kernels -----------------------------

__global__ void cvt_f32_bf16_kernel(const float* __restrict__ in,
                                    __bf16* __restrict__ out, int n) {
  int i = blockIdx.x * blockDim.x + threadIdx.x;
  if (i < n) out[i] = (__bf16)in[i];
}

// w (d, n, h) fp32 -> wb[(n*64+h)][d] bf16  (K = d contiguous per output col)
__global__ void cvt_w_qkv_kernel(const float* __restrict__ in,
                                 __bf16* __restrict__ out) {
  int i = blockIdx.x * blockDim.x + threadIdx.x;   // over 1024*1024
  int d = i >> 10;
  int c = i & 1023;                                // c = n*64+h
  out[(size_t)c * DM + d] = (__bf16)in[i];
}

// wp (n, d, h) fp32 -> wpb[d][(n*64+h)] bf16  (K = n*64+h contiguous per col d)
__global__ void cvt_w_proj_kernel(const float* __restrict__ in,
                                  __bf16* __restrict__ out) {
  int i = blockIdx.x * blockDim.x + threadIdx.x;   // over 16*1024*64
  int n = i >> 16;
  int rem = i & 65535;
  int d = rem >> 6;
  int h = rem & 63;
  out[(size_t)d * (NH * H) + n * H + h] = (__bf16)in[i];
}

// ----------------------------- QKV GEMM ------------------------------------
// C[bt, h] = sum_d x[bt,d] * w[d, head, h] + bias[head,h]
// out layout: transposed==0 -> [b][head][t][h] ; transposed==1 -> [b][head][h][t]

__global__ __launch_bounds__(256) void qkv_gemm_kernel(
    const __bf16* __restrict__ xb,    // [BT][DM]
    const __bf16* __restrict__ wb,    // [NH*H][DM]  (K-major per col)
    const float*  __restrict__ bias,  // [NH][1][H]
    __bf16* __restrict__ out,
    int transposed) {
  const int lane = threadIdx.x & 31;
  const int wave = threadIdx.x >> 5;
  const int half = lane >> 4;
  const int l16  = lane & 15;
  const int head = blockIdx.y;
  const int row0 = (blockIdx.x * 8 + wave) * 16;

  v8f acc[4] = {};
  const __bf16* arow  = xb + (size_t)(row0 + l16) * DM;
  const __bf16* wbase = wb + (size_t)head * H * DM;

  for (int kk = 0; kk < DM; kk += 32) {
    __builtin_prefetch((const void*)(arow + kk + 256), 0, 1);
    v8bf alo = *(const v8bf*)(arow + kk + half * 16);
    v8bf ahi = *(const v8bf*)(arow + kk + half * 16 + 8);
    v16bf afrag = combine8(alo, ahi);
#pragma unroll
    for (int c = 0; c < 4; ++c) {
      int ncol = c * 16 + l16;
      v16bf bfrag = *(const v16bf*)(wbase + (size_t)ncol * DM + kk + half * 16);
      acc[c] = WMMA_BF16(afrag, bfrag, acc[c]);
    }
  }
#pragma unroll
  for (int c = 0; c < 4; ++c) {
    int hcol = c * 16 + l16;
    float bv = bias[head * H + hcol];
#pragma unroll
    for (int r = 0; r < 8; ++r) {
      int bt = row0 + r + 8 * half;
      int bb = bt >> 11, tt = bt & (T - 1);
      float v = acc[c][r] + bv;
      size_t o = transposed
        ? ((size_t)(bb * NH + head) * H + hcol) * T + tt
        : ((size_t)(bb * NH + head) * T + tt) * H + hcol;
      out[o] = (__bf16)v;
    }
  }
}

// --------------------------- flash attention -------------------------------
// Per wave: 16 query rows x full h=64, online softmax over 32-key blocks.
// K/V blocks are staged into double-buffered LDS by the Tensor Data Mover
// (one issuing wave, TENSORcnt + workgroup barrier for publication).

__global__ __launch_bounds__(256) void attn_kernel(
    const __bf16* __restrict__ Q,   // [b][n][t][h]
    const __bf16* __restrict__ K,   // [b][n][t][h]
    const __bf16* __restrict__ Vt,  // [b][n][h][t]
    __bf16* __restrict__ heads) {   // [b][t][n*64+h]
  __shared__ __align__(16) __bf16 KB[2][32][H];   // 32 keys x 64 (8 KiB)
  __shared__ __align__(16) __bf16 VB[2][H][32];   // 64 h    x 32 (8 KiB)
  __shared__ __align__(16) __bf16 plds[8][16][32];

  const int lane = threadIdx.x & 31;
  const int wave = threadIdx.x >> 5;
  const int half = lane >> 4;
  const int l16  = lane & 15;
  const int head = blockIdx.y;
  const int bb   = blockIdx.z;
  const int q0   = (blockIdx.x * 8 + wave) * 16;

  const size_t bh = (size_t)(bb * NH + head);
  const __bf16* Qb = Q  + bh * T * H;
  const __bf16* Kb = K  + bh * T * H;
  const __bf16* Vb = Vt + bh * H * T;

  // Q fragments (A layout) for the two 32-wide k-steps over h
  const __bf16* qrow = Qb + (size_t)(q0 + l16) * H;
  v16bf qf[2];
#pragma unroll
  for (int s = 0; s < 2; ++s) {
    v8bf lo = *(const v8bf*)(qrow + s * 32 + half * 16);
    v8bf hi = *(const v8bf*)(qrow + s * 32 + half * 16 + 8);
    qf[s] = combine8(lo, hi);
  }

  v8f O[4] = {};
  float mrow[8], lrow[8];
#pragma unroll
  for (int r = 0; r < 8; ++r) { mrow[r] = -1e30f; lrow[r] = 0.f; }

  // TDM prologue: stage block 0 into buffer 0
  if (wave == 0) {
    tdm_load_2d(Kb, &KB[0][0][0], /*tile*/ H, 32, /*tensor*/ H, T, /*stride0*/ H);
    tdm_load_2d(Vb, &VB[0][0][0], /*tile*/ 32, H, /*tensor*/ T, H, /*stride0*/ T);
  }

  int p = 0;
  for (int jb = 0; jb < T; jb += 32, p ^= 1) {
    if (wave == 0) __builtin_amdgcn_s_wait_tensorcnt(0);
    __syncthreads();                       // publish buffer p to all waves
    if (wave == 0 && jb + 32 < T) {        // prefetch next block into p^1
      tdm_load_2d(Kb + (size_t)(jb + 32) * H, &KB[p ^ 1][0][0], H, 32, H, T, H);
      tdm_load_2d(Vb + (size_t)(jb + 32),     &VB[p ^ 1][0][0], 32, H, T, H, T);
    }

    // S = Q * K^T for 32 keys: two 16x16 tiles, B-fragments from LDS
    v8f s0 = {}, s1 = {};
#pragma unroll
    for (int s = 0; s < 2; ++s) {
      v16bf k0 = *(const v16bf*)&KB[p][l16]     [s * 32 + half * 16];
      v16bf k1 = *(const v16bf*)&KB[p][16 + l16][s * 32 + half * 16];
      s0 = WMMA_BF16(qf[s], k0, s0);
      s1 = WMMA_BF16(qf[s], k1, s1);
    }
    // online softmax per row (rows M = r + 8*half striped over 16-lane halves)
    float p0[8], p1[8];
#pragma unroll
    for (int r = 0; r < 8; ++r) {
      float a = s0[r] * SCALE;
      float c = s1[r] * SCALE;
      float mx = fmaxf(a, c);
#pragma unroll
      for (int d = 1; d < 16; d <<= 1) mx = fmaxf(mx, __shfl_xor(mx, d, 32));
      float mnew  = fmaxf(mrow[r], mx);
      float alpha = __expf(mrow[r] - mnew);
      float e0 = __expf(a - mnew);
      float e1 = __expf(c - mnew);
      float sum = e0 + e1;
#pragma unroll
      for (int d = 1; d < 16; d <<= 1) sum += __shfl_xor(sum, d, 32);
      lrow[r] = lrow[r] * alpha + sum;
      mrow[r] = mnew;
      p0[r] = e0; p1[r] = e1;
#pragma unroll
      for (int c2 = 0; c2 < 4; ++c2) O[c2][r] *= alpha;
    }
    // C-layout -> A-layout conversion via wave-private LDS tile (in-order DS)
#pragma unroll
    for (int r = 0; r < 8; ++r) {
      plds[wave][r + 8 * half][l16]      = (__bf16)p0[r];
      plds[wave][r + 8 * half][16 + l16] = (__bf16)p1[r];
    }
    asm volatile("s_wait_dscnt 0x0" ::: "memory");
    v8bf plo = *(const v8bf*)&plds[wave][l16][half * 16];
    v8bf phi = *(const v8bf*)&plds[wave][l16][half * 16 + 8];
    v16bf pf = combine8(plo, phi);
    // O += P * V, V-fragments from LDS (rows of VB are key-contiguous)
#pragma unroll
    for (int c = 0; c < 4; ++c) {
      int hcol = c * 16 + l16;
      v16bf vf = *(const v16bf*)&VB[p][hcol][half * 16];
      O[c] = WMMA_BF16(pf, vf, O[c]);
    }
  }
  // normalize rows and store heads (b, t, n*64+h) bf16 for the projection GEMM
#pragma unroll
  for (int r = 0; r < 8; ++r) {
    float inv = 1.f / lrow[r];
    int t = q0 + r + 8 * half;
    __bf16* hrow = heads + ((size_t)(bb * T + t) * (NH * H)) + head * H;
#pragma unroll
    for (int c = 0; c < 4; ++c) hrow[c * 16 + l16] = (__bf16)(O[c][r] * inv);
  }
}

// --------------------------- output projection -----------------------------
// out[bt, d] = sum_k heads[bt, k] * wpb[d, k] + bias[d]

__global__ __launch_bounds__(256) void proj_gemm_kernel(
    const __bf16* __restrict__ heads,  // [BT][1024]
    const __bf16* __restrict__ wpb,    // [1024 dcol][1024 k]
    const float*  __restrict__ bias,   // [1024]
    float* __restrict__ out) {         // [BT][1024]
  const int lane = threadIdx.x & 31;
  const int wave = threadIdx.x >> 5;
  const int half = lane >> 4;
  const int l16  = lane & 15;
  const int row0 = (blockIdx.x * 8 + wave) * 16;
  const int col0 = blockIdx.y * 64;

  v8f acc[4] = {};
  const __bf16* arow = heads + (size_t)(row0 + l16) * (NH * H);

  for (int kk = 0; kk < NH * H; kk += 32) {
    __builtin_prefetch((const void*)(arow + kk + 256), 0, 1);
    v8bf alo = *(const v8bf*)(arow + kk + half * 16);
    v8bf ahi = *(const v8bf*)(arow + kk + half * 16 + 8);
    v16bf afrag = combine8(alo, ahi);
#pragma unroll
    for (int c = 0; c < 4; ++c) {
      int ncol = col0 + c * 16 + l16;
      v16bf bfrag = *(const v16bf*)(wpb + (size_t)ncol * (NH * H) + kk + half * 16);
      acc[c] = WMMA_BF16(afrag, bfrag, acc[c]);
    }
  }
#pragma unroll
  for (int c = 0; c < 4; ++c) {
    int dcol = col0 + c * 16 + l16;
    float bv = bias[dcol];
#pragma unroll
    for (int r = 0; r < 8; ++r) {
      int bt = row0 + r + 8 * half;
      out[(size_t)bt * DM + dcol] = acc[c][r] + bv;
    }
  }
}

// ------------------------------- launcher ----------------------------------

extern "C" void kernel_launch(void* const* d_in, const int* in_sizes, int n_in,
                              void* d_out, int out_size, void* d_ws, size_t ws_size,
                              hipStream_t stream) {
  const float* x  = (const float*)d_in[0];
  const float* wq = (const float*)d_in[1];
  const float* bq = (const float*)d_in[2];
  const float* wk = (const float*)d_in[3];
  const float* bk = (const float*)d_in[4];
  const float* wv = (const float*)d_in[5];
  const float* bv = (const float*)d_in[6];
  const float* wp = (const float*)d_in[7];
  const float* bp = (const float*)d_in[8];
  float* out = (float*)d_out;

  // workspace carve-up (all bf16), 48 MiB total
  char* ws = (char*)d_ws;
  __bf16* xb    = (__bf16*)ws; ws += (size_t)BT * DM * 2;        // 8 MiB
  __bf16* wqb   = (__bf16*)ws; ws += (size_t)DM * NH * H * 2;    // 2 MiB
  __bf16* wkb   = (__bf16*)ws; ws += (size_t)DM * NH * H * 2;    // 2 MiB
  __bf16* wvb   = (__bf16*)ws; ws += (size_t)DM * NH * H * 2;    // 2 MiB
  __bf16* wpb   = (__bf16*)ws; ws += (size_t)DM * NH * H * 2;    // 2 MiB
  __bf16* Qbuf  = (__bf16*)ws; ws += (size_t)B * NH * T * H * 2; // 8 MiB
  __bf16* Kbuf  = (__bf16*)ws; ws += (size_t)B * NH * T * H * 2; // 8 MiB
  __bf16* Vtbuf = (__bf16*)ws; ws += (size_t)B * NH * T * H * 2; // 8 MiB
  __bf16* hbuf  = (__bf16*)ws; ws += (size_t)BT * NH * H * 2;    // 8 MiB

  // 1) conversions / relayouts
  {
    int n = BT * DM;
    cvt_f32_bf16_kernel<<<(n + 255) / 256, 256, 0, stream>>>(x, xb, n);
  }
  cvt_w_qkv_kernel<<<(DM * NH * H) / 256, 256, 0, stream>>>(wq, wqb);
  cvt_w_qkv_kernel<<<(DM * NH * H) / 256, 256, 0, stream>>>(wk, wkb);
  cvt_w_qkv_kernel<<<(DM * NH * H) / 256, 256, 0, stream>>>(wv, wvb);
  cvt_w_proj_kernel<<<(NH * DM * H) / 256, 256, 0, stream>>>(wp, wpb);

  // 2) QKV projections (V written transposed for P*V fragment loads)
  dim3 g_qkv(BT / 128, NH);
  qkv_gemm_kernel<<<g_qkv, 256, 0, stream>>>(xb, wqb, bq, Qbuf, 0);
  qkv_gemm_kernel<<<g_qkv, 256, 0, stream>>>(xb, wkb, bk, Kbuf, 0);
  qkv_gemm_kernel<<<g_qkv, 256, 0, stream>>>(xb, wvb, bv, Vtbuf, 1);

  // 3) flash attention (TDM-staged K/V)
  dim3 g_attn(T / 128, NH, B);
  attn_kernel<<<g_attn, 256, 0, stream>>>(Qbuf, Kbuf, Vtbuf, hbuf);

  // 4) output projection + bias
  dim3 g_proj(BT / 128, DM / 64);
  proj_gemm_kernel<<<g_proj, 256, 0, stream>>>(hbuf, wpb, bp, out);
}
